// AttentionClassifier_82360292868112
// MI455X (gfx1250) — matrix operand order repaired and checked by reading, not compiled, not run on hardware
//
#include <hip/hip_runtime.h>
#include <hip/hip_bf16.h>

// ---------------------------------------------------------------------------
// Problem dims
// ---------------------------------------------------------------------------
#define B_SZ   512
#define T_LEN  200
#define E_DIM  100
#define E_P    128    // embedding K padded to mult of 32
#define H1_DIM 100
#define H1_P   128    // hidden1 K padded
#define G1_P   304    // 3*H1 gates padded to mult of 16
#define CAT    400    // concat(h1, x_hat)
#define CAT_P  416    // padded (mult of 32)
#define H2_DIM 400
#define H2_P   416
#define G2_P   1200   // 3*H2, already mult of 16
#define D2     800    // 2*H2
#define SROW_P 224    // attention row padded (200 -> 224)

typedef __attribute__((ext_vector_type(16))) _Float16 v16h;
typedef __attribute__((ext_vector_type(8)))  float    v8f;

// ---------------------------------------------------------------------------
// WMMA fragment helpers (wave32, v_wmma_f32_16x16x32_f16 layouts, ISA 7.12.2)
// A (16x32 f16): lanes 0-15 M=0..15 hold K{0..7,16..23}; lanes 16-31 K{8..15,24..31}
// B (32x16 f16): lane&15 = N; lanes 0-15 hold K{0..7,16..23}; lanes 16-31 K{8..15,24..31}
// C/D (16x16 f32): v8f, element r -> M = r + 8*(lane>>4), N = lane&15
// ---------------------------------------------------------------------------
__device__ inline v16h wmma_load_a(const _Float16* __restrict__ base, int strideM,
                                   int mLimit, int kLimit) {
  const int lane = threadIdx.x & 31;
  const int m = lane & 15;
  const int hi = lane >> 4;
  v16h a = {};
  if (m < mLimit) {
#pragma unroll
    for (int v = 0; v < 8; ++v) {
      int k = (v < 4 ? 2 * v : 16 + 2 * (v - 4)) + (hi ? 8 : 0);
      a[2 * v]     = (k     < kLimit) ? base[m * strideM + k]     : (_Float16)0.f;
      a[2 * v + 1] = (k + 1 < kLimit) ? base[m * strideM + k + 1] : (_Float16)0.f;
    }
  }
  return a;
}

__device__ inline v16h wmma_load_b(const _Float16* __restrict__ base, int strideK,
                                   int strideN, int nLimit, int kLimit) {
  const int lane = threadIdx.x & 31;
  const int n = lane & 15;
  const int hi = lane >> 4;
  v16h b = {};
  if (n < nLimit) {
#pragma unroll
    for (int v = 0; v < 8; ++v) {
      int k = (v < 4 ? 2 * v : 16 + 2 * (v - 4)) + (hi ? 8 : 0);
      b[2 * v]     = (k     < kLimit) ? base[k * strideK + n * strideN]       : (_Float16)0.f;
      b[2 * v + 1] = (k + 1 < kLimit) ? base[(k + 1) * strideK + n * strideN] : (_Float16)0.f;
    }
  }
  return b;
}

__device__ inline void wmma_store_f32(float* __restrict__ base, int strideM, v8f c,
                                      int mLimit, int nLimit) {
  const int lane = threadIdx.x & 31;
  const int n = lane & 15;
  const int hi = lane >> 4;
  if (n < nLimit) {
#pragma unroll
    for (int r = 0; r < 8; ++r) {
      int m = r + 8 * hi;
      if (m < mLimit) base[m * strideM + n] = c[r];
    }
  }
}

__device__ inline void wmma_store_f16(_Float16* __restrict__ base, int strideM, v8f c,
                                      int mLimit, int nLimit) {
  const int lane = threadIdx.x & 31;
  const int n = lane & 15;
  const int hi = lane >> 4;
  if (n < nLimit) {
#pragma unroll
    for (int r = 0; r < 8; ++r) {
      int m = r + 8 * hi;
      if (m < mLimit) base[m * strideM + n] = (_Float16)c[r];
    }
  }
}

#define WMMA_F32_F16(a, b, c) \
  __builtin_amdgcn_wmma_f32_16x16x32_f16(false, (a), false, (b), (short)0, (c), false, false)

// ---------------------------------------------------------------------------
// Weight conversion: f32 (N,K) row-major -> f16 (Np,Kp) zero-padded
// ---------------------------------------------------------------------------
__global__ void convert_weight_kernel(const float* __restrict__ src, _Float16* __restrict__ dst,
                                      int N, int K, int Kp, int total) {
  int idx = blockIdx.x * 256 + threadIdx.x;
  if (idx >= total) return;
  int n = idx / Kp, k = idx % Kp;
  float v = (n < N && k < K) ? src[n * K + k] : 0.f;
  dst[idx] = (_Float16)v;
}

// ---------------------------------------------------------------------------
// Embedding gather -> f16 padded (B,T,E_P)
// ---------------------------------------------------------------------------
__global__ void embed_kernel(const int* __restrict__ x, const float* __restrict__ emb,
                             _Float16* __restrict__ e_h) {
  size_t idx = (size_t)blockIdx.x * 256 + threadIdx.x;
  if (idx >= (size_t)B_SZ * T_LEN * E_P) return;
  int k = (int)(idx & (E_P - 1));
  size_t bt = idx >> 7;
  int tok = x[bt];
  e_h[idx] = (k < E_DIM) ? (_Float16)emb[(size_t)tok * E_DIM + k] : (_Float16)0.f;
}

// ---------------------------------------------------------------------------
// Persistent GRU scan. One block = 16 batch rows, loops over all timesteps.
// Per step: xg = x_t @ Wih^T (WMMA), hg = h @ Whh^T (WMMA), then gate math.
// KP: padded input K; KLIM: valid input K; H: hidden; HP: padded hidden K;
// GP: padded gate count (mult of 16).
// ---------------------------------------------------------------------------
template <int KP, int KLIM, int H, int HP, int GP>
__global__ void gru_scan_kernel(const _Float16* __restrict__ xin, int xin_stride,
                                const _Float16* __restrict__ Wih,
                                const _Float16* __restrict__ Whh,
                                const float* __restrict__ bih, const float* __restrict__ bhh,
                                _Float16* __restrict__ out, int out_stride, int out_col,
                                int dir) {
  extern __shared__ char smem[];
  float*    xg  = (float*)smem;                 // 16*GP
  float*    hg  = xg + 16 * GP;                 // 16*GP
  float*    h32 = hg + 16 * GP;                 // 16*H
  _Float16* h16 = (_Float16*)(h32 + 16 * H);    // 16*HP (zero-padded cols)

  const int tid = threadIdx.x;
  const int wave = tid >> 5;
  const int nwaves = blockDim.x >> 5;
  const int b0 = blockIdx.x * 16;
  const int rowStrideA = T_LEN * xin_stride;    // batch-row stride of xin

  for (int i = tid; i < 16 * H; i += blockDim.x) h32[i] = 0.f;
  for (int i = tid; i < 16 * HP; i += blockDim.x) h16[i] = (_Float16)0.f;
  __syncthreads();

  for (int t = 0; t < T_LEN; ++t) {
    const int tt = dir ? (T_LEN - 1 - t) : t;
    const _Float16* xrow = xin + ((size_t)b0 * T_LEN + tt) * xin_stride;

    for (int nt = wave; nt < GP / 16; nt += nwaves) {
      v8f cx = {};
      v8f ch = {};
      for (int kk = 0; kk < KP / 32; ++kk) {
        v16h a = wmma_load_a(xrow + kk * 32, rowStrideA, 16, KLIM - kk * 32);
        v16h b = wmma_load_b(Wih + (size_t)(nt * 16) * KP + kk * 32, 1, KP, 16, 32);
        cx = WMMA_F32_F16(a, b, cx);
      }
      for (int kk = 0; kk < HP / 32; ++kk) {
        v16h a = wmma_load_a(h16 + kk * 32, HP, 16, 32);
        v16h b = wmma_load_b(Whh + (size_t)(nt * 16) * HP + kk * 32, 1, HP, 16, 32);
        ch = WMMA_F32_F16(a, b, ch);
      }
      wmma_store_f32(xg + nt * 16, GP, cx, 16, 16);
      wmma_store_f32(hg + nt * 16, GP, ch, 16, 16);
    }
    __syncthreads();

    for (int i = tid; i < 16 * H; i += blockDim.x) {
      int m = i / H, j = i % H;
      float xr = xg[m * GP + j]         + bih[j];
      float xz = xg[m * GP + H + j]     + bih[H + j];
      float xn = xg[m * GP + 2 * H + j] + bih[2 * H + j];
      float hr = hg[m * GP + j]         + bhh[j];
      float hz = hg[m * GP + H + j]     + bhh[H + j];
      float hn = hg[m * GP + 2 * H + j] + bhh[2 * H + j];
      float r = 1.f / (1.f + __expf(-(xr + hr)));
      float z = 1.f / (1.f + __expf(-(xz + hz)));
      float n = tanhf(xn + r * hn);
      float hv = (1.f - z) * n + z * h32[i];
      h32[i] = hv;
      h16[m * HP + j] = (_Float16)hv;
      out[((size_t)(b0 + m) * T_LEN + tt) * out_stride + out_col + j] = (_Float16)hv;
    }
    __syncthreads();
  }
}

// ---------------------------------------------------------------------------
// s[b,t,u] = dot(h1[b,t,:], h1[b,u,:])  (K=200, h1 = cat cols [0,200))
// ---------------------------------------------------------------------------
__global__ __launch_bounds__(32) void attn_scores_kernel(const _Float16* __restrict__ cat,
                                                         float* __restrict__ s_f) {
  const int nt = blockIdx.x, mt = blockIdx.y, b = blockIdx.z;
  const _Float16* h1 = cat + (size_t)b * T_LEN * CAT_P;
  int mLim = T_LEN - mt * 16; if (mLim > 16) mLim = 16;
  int nLim = T_LEN - nt * 16; if (nLim > 16) nLim = 16;
  v8f c = {};
  for (int kk = 0; kk < 7; ++kk) {
    int k0 = kk * 32;
    v16h a = wmma_load_a(h1 + (size_t)(mt * 16) * CAT_P + k0, CAT_P, mLim, 200 - k0);
    v16h bf = wmma_load_b(h1 + (size_t)(nt * 16) * CAT_P + k0, 1, CAT_P, nLim, 200 - k0);
    c = WMMA_F32_F16(a, bf, c);
  }
  wmma_store_f32(s_f + ((size_t)b * T_LEN + mt * 16) * T_LEN + nt * 16, T_LEN, c, mLim, nLim);
}

// ---------------------------------------------------------------------------
// Row softmax: zero-diag logits -> softmax -> zero diag -> mask -> renorm+EPS
// One wave per row. Output f16 padded to SROW_P.
// ---------------------------------------------------------------------------
__global__ __launch_bounds__(256) void attn_softmax_kernel(const float* __restrict__ s_f,
                                                           const int* __restrict__ mask,
                                                           _Float16* __restrict__ s_h) {
  const int row = blockIdx.x * 8 + (threadIdx.x >> 5);
  const int lane = threadIdx.x & 31;
  const int b = row / T_LEN, t = row % T_LEN;
  const float* sr = s_f + ((size_t)b * T_LEN + t) * T_LEN;
  float vals[7];
  float mx = -3.4e38f;
#pragma unroll
  for (int i = 0; i < 7; ++i) {
    int u = lane + 32 * i;
    float v = -3.4e38f;
    if (u < T_LEN) v = (u == t) ? 0.f : sr[u];
    vals[i] = v;
    mx = fmaxf(mx, v);
  }
  for (int o = 16; o; o >>= 1) mx = fmaxf(mx, __shfl_xor(mx, o, 32));
  float s = 0.f;
#pragma unroll
  for (int i = 0; i < 7; ++i) {
    int u = lane + 32 * i;
    if (u < T_LEN) { float p = __expf(vals[i] - mx); vals[i] = p; s += p; }
  }
  for (int o = 16; o; o >>= 1) s += __shfl_xor(s, o, 32);
  const float mt = (float)mask[b * T_LEN + t];
  float s2 = 0.f;
#pragma unroll
  for (int i = 0; i < 7; ++i) {
    int u = lane + 32 * i;
    if (u < T_LEN) {
      float p = vals[i] / s;
      if (u == t) p = 0.f;
      p *= mt * (float)mask[b * T_LEN + u];
      vals[i] = p;
      s2 += p;
    }
  }
  for (int o = 16; o; o >>= 1) s2 += __shfl_xor(s2, o, 32);
  const float inv = 1.f / (s2 + 1e-13f);
  _Float16* dst = s_h + ((size_t)b * T_LEN + t) * SROW_P;
#pragma unroll
  for (int i = 0; i < 7; ++i) {
    int u = lane + 32 * i;
    if (u < SROW_P) dst[u] = (u < T_LEN) ? (_Float16)(vals[i] * inv) : (_Float16)0.f;
  }
}

// ---------------------------------------------------------------------------
// x_hat = s @ h1 -> written into cat cols [200,400)
// ---------------------------------------------------------------------------
__global__ __launch_bounds__(32) void attn_xhat_kernel(const _Float16* __restrict__ s_h,
                                                       _Float16* __restrict__ cat) {
  const int nt = blockIdx.x, mt = blockIdx.y, b = blockIdx.z;
  int mLim = T_LEN - mt * 16; if (mLim > 16) mLim = 16;
  int nLim = T_LEN - nt * 16; if (nLim > 16) nLim = 16;
  const _Float16* srow = s_h + (size_t)b * T_LEN * SROW_P;
  const _Float16* h1 = cat + (size_t)b * T_LEN * CAT_P;
  v8f c = {};
  for (int kk = 0; kk < 7; ++kk) {
    int k0 = kk * 32;
    v16h a = wmma_load_a(srow + (size_t)(mt * 16) * SROW_P + k0, SROW_P, mLim, 32);
    v16h bf = wmma_load_b(h1 + (size_t)k0 * CAT_P + nt * 16, CAT_P, 1, nLim, 200 - k0);
    c = WMMA_F32_F16(a, bf, c);
  }
  wmma_store_f16(cat + ((size_t)b * T_LEN + (size_t)mt * 16) * CAT_P + 200 + nt * 16,
                 CAT_P, c, mLim, nLim);
}

// ---------------------------------------------------------------------------
// Attention pooling: logits = h2 . ctx, masked softmax over T, weighted sum.
// ---------------------------------------------------------------------------
__global__ __launch_bounds__(256) void attn_pool_kernel(const _Float16* __restrict__ h2,
                                                        const float* __restrict__ ctx,
                                                        const int* __restrict__ mask,
                                                        float* __restrict__ pooled) {
  const int b = blockIdx.x;
  __shared__ float w[T_LEN];
  __shared__ float inv_s;
  const int tid = threadIdx.x;
  if (tid < T_LEN) {
    const _Float16* hrow = h2 + ((size_t)b * T_LEN + tid) * D2;
    float acc = 0.f;
    for (int j = 0; j < D2; ++j) acc += (float)hrow[j] * ctx[j];
    w[tid] = acc;
  }
  __syncthreads();
  if (tid < 32) {
    float mx = -3.4e38f;
    for (int t2 = tid; t2 < T_LEN; t2 += 32) mx = fmaxf(mx, w[t2]);
    for (int o = 16; o; o >>= 1) mx = fmaxf(mx, __shfl_xor(mx, o, 32));
    float s = 0.f;
    for (int t2 = tid; t2 < T_LEN; t2 += 32) s += __expf(w[t2] - mx);
    for (int o = 16; o; o >>= 1) s += __shfl_xor(s, o, 32);
    float s2 = 0.f;
    for (int t2 = tid; t2 < T_LEN; t2 += 32) {
      float a = __expf(w[t2] - mx) / s * (float)mask[b * T_LEN + t2];
      w[t2] = a;
      s2 += a;
    }
    for (int o = 16; o; o >>= 1) s2 += __shfl_xor(s2, o, 32);
    if (tid == 0) inv_s = 1.f / (s2 + 1e-13f);
  }
  __syncthreads();
  const float inv = inv_s;
  for (int j = tid; j < D2; j += 256) {
    float acc = 0.f;
    for (int t2 = 0; t2 < T_LEN; ++t2)
      acc += w[t2] * (float)h2[((size_t)b * T_LEN + t2) * D2 + j];
    pooled[(size_t)b * D2 + j] = acc * inv;
  }
}

// ---------------------------------------------------------------------------
// out = pooled @ final_W^T + final_b
// ---------------------------------------------------------------------------
__global__ __launch_bounds__(32) void final_fc_kernel(const float* __restrict__ pooled,
                                                      const float* __restrict__ W,
                                                      const float* __restrict__ bias,
                                                      float* __restrict__ out) {
  const int b = blockIdx.x, c = threadIdx.x;
  if (c < 20) {
    const float* p = pooled + (size_t)b * D2;
    const float* wr = W + (size_t)c * D2;
    float acc = bias[c];
    for (int j = 0; j < D2; ++j) acc += p[j] * wr[j];
    out[b * 20 + c] = acc;
  }
}

// ---------------------------------------------------------------------------
// Host launcher
// ---------------------------------------------------------------------------
extern "C" void kernel_launch(void* const* d_in, const int* in_sizes, int n_in,
                              void* d_out, int out_size, void* d_ws, size_t ws_size,
                              hipStream_t stream) {
  (void)in_sizes; (void)n_in; (void)out_size; (void)ws_size;
  const int*   x       = (const int*)d_in[0];
  const int*   mask    = (const int*)d_in[2];
  const float* emb     = (const float*)d_in[3];
  const float* g1f_Wih = (const float*)d_in[4];
  const float* g1f_Whh = (const float*)d_in[5];
  const float* g1f_bih = (const float*)d_in[6];
  const float* g1f_bhh = (const float*)d_in[7];
  const float* g1b_Wih = (const float*)d_in[8];
  const float* g1b_Whh = (const float*)d_in[9];
  const float* g1b_bih = (const float*)d_in[10];
  const float* g1b_bhh = (const float*)d_in[11];
  const float* g2f_Wih = (const float*)d_in[12];
  const float* g2f_Whh = (const float*)d_in[13];
  const float* g2f_bih = (const float*)d_in[14];
  const float* g2f_bhh = (const float*)d_in[15];
  const float* g2b_Wih = (const float*)d_in[16];
  const float* g2b_Whh = (const float*)d_in[17];
  const float* g2b_bih = (const float*)d_in[18];
  const float* g2b_bhh = (const float*)d_in[19];
  const float* ctx     = (const float*)d_in[20];
  const float* finW    = (const float*)d_in[21];
  const float* finB    = (const float*)d_in[22];

  char* ws = (char*)d_ws;
  size_t off = 0;
  auto take = [&](size_t bytes) -> char* {
    char* p = ws + off;
    off += (bytes + 255) & ~(size_t)255;
    return p;
  };
  _Float16* W1f_ih = (_Float16*)take((size_t)G1_P * E_P * 2);
  _Float16* W1f_hh = (_Float16*)take((size_t)G1_P * H1_P * 2);
  _Float16* W1b_ih = (_Float16*)take((size_t)G1_P * E_P * 2);
  _Float16* W1b_hh = (_Float16*)take((size_t)G1_P * H1_P * 2);
  _Float16* W2f_ih = (_Float16*)take((size_t)G2_P * CAT_P * 2);
  _Float16* W2f_hh = (_Float16*)take((size_t)G2_P * H2_P * 2);
  _Float16* W2b_ih = (_Float16*)take((size_t)G2_P * CAT_P * 2);
  _Float16* W2b_hh = (_Float16*)take((size_t)G2_P * H2_P * 2);
  _Float16* e_h    = (_Float16*)take((size_t)B_SZ * T_LEN * E_P * 2);
  _Float16* cat    = (_Float16*)take((size_t)B_SZ * T_LEN * CAT_P * 2);
  float*    s_f    = (float*)take((size_t)B_SZ * T_LEN * T_LEN * 4);
  _Float16* s_h    = (_Float16*)take((size_t)B_SZ * T_LEN * SROW_P * 2);
  _Float16* h2     = (_Float16*)take((size_t)B_SZ * T_LEN * D2 * 2);
  float*    pooled = (float*)take((size_t)B_SZ * D2 * 4);

  auto convw = [&](const float* src, _Float16* dst, int N, int K, int Np, int Kp) {
    int tot = Np * Kp;
    convert_weight_kernel<<<(tot + 255) / 256, 256, 0, stream>>>(src, dst, N, K, Kp, tot);
  };
  convw(g1f_Wih, W1f_ih, 300, 100, G1_P, E_P);
  convw(g1f_Whh, W1f_hh, 300, 100, G1_P, H1_P);
  convw(g1b_Wih, W1b_ih, 300, 100, G1_P, E_P);
  convw(g1b_Whh, W1b_hh, 300, 100, G1_P, H1_P);
  convw(g2f_Wih, W2f_ih, 1200, 400, G2_P, CAT_P);
  convw(g2f_Whh, W2f_hh, 1200, 400, G2_P, H2_P);
  convw(g2b_Wih, W2b_ih, 1200, 400, G2_P, CAT_P);
  convw(g2b_Whh, W2b_hh, 1200, 400, G2_P, H2_P);

  {
    size_t tot = (size_t)B_SZ * T_LEN * E_P;
    embed_kernel<<<(unsigned)((tot + 255) / 256), 256, 0, stream>>>(x, emb, e_h);
  }

  // Layer-1 BiGRU: 32 persistent blocks (16 batch rows each), 8 waves.
  size_t lds1 = (size_t)16 * G1_P * 4 * 2 + 16 * H1_DIM * 4 + 16 * H1_P * 2;
  gru_scan_kernel<E_P, E_P, H1_DIM, H1_P, G1_P><<<B_SZ / 16, 256, lds1, stream>>>(
      e_h, E_P, W1f_ih, W1f_hh, g1f_bih, g1f_bhh, cat, CAT_P, 0, 0);
  gru_scan_kernel<E_P, E_P, H1_DIM, H1_P, G1_P><<<B_SZ / 16, 256, lds1, stream>>>(
      e_h, E_P, W1b_ih, W1b_hh, g1b_bih, g1b_bhh, cat, CAT_P, H1_DIM, 1);

  // Self-attention
  attn_scores_kernel<<<dim3(13, 13, B_SZ), 32, 0, stream>>>(cat, s_f);
  attn_softmax_kernel<<<(B_SZ * T_LEN) / 8, 256, 0, stream>>>(s_f, mask, s_h);
  attn_xhat_kernel<<<dim3(13, 13, B_SZ), 32, 0, stream>>>(s_h, cat);

  // Layer-2 BiGRU: 32 persistent blocks, 16 waves, ~188KB dynamic LDS (WGP has 320KB).
  size_t lds2 = (size_t)16 * G2_P * 4 * 2 + 16 * H2_DIM * 4 + 16 * H2_P * 2;
  gru_scan_kernel<CAT_P, CAT, H2_DIM, H2_P, G2_P><<<B_SZ / 16, 512, lds2, stream>>>(
      cat, CAT_P, W2f_ih, W2f_hh, g2f_bih, g2f_bhh, h2, D2, 0, 0);
  gru_scan_kernel<CAT_P, CAT, H2_DIM, H2_P, G2_P><<<B_SZ / 16, 512, lds2, stream>>>(
      cat, CAT_P, W2b_ih, W2b_hh, g2b_bih, g2b_bhh, h2, D2, H2_DIM, 1);

  // Pool + classify
  attn_pool_kernel<<<B_SZ, 256, 0, stream>>>(h2, ctx, mask, pooled);
  final_fc_kernel<<<B_SZ, 32, 0, stream>>>(pooled, finW, finB, (float*)d_out);
}
